// SimilarityTripletLoss_10642928959948
// MI455X (gfx1250) — matrix operand order repaired and checked by reading, not compiled, not run on hardware
//
#include <hip/hip_runtime.h>
#include <hip/hip_bf16.h>
#include <math.h>

#define BATCH 4
#define CDIM 256
#define WDIM 64
#define MDIM 4096            // H*W, also N (pairs per batch)
#define EPS 1e-8f
#define MARGIN 0.6f
#define POS_MASK 2.0f
#define FLT_BIG 3.0e38f

typedef __attribute__((ext_vector_type(16))) __bf16 v16bf;
typedef __attribute__((ext_vector_type(8)))  __bf16 v8bf;
typedef __attribute__((ext_vector_type(8)))  unsigned short v8us;
typedef __attribute__((ext_vector_type(8)))  float v8f;

union U8 { v8us u; v8bf b; };

// float -> bf16 round-to-nearest-even
static __device__ __forceinline__ unsigned short f2bf(float f) {
  unsigned u = __float_as_uint(f);
  unsigned r = u + 0x7FFFu + ((u >> 16) & 1u);
  return (unsigned short)(r >> 16);
}
static __device__ __forceinline__ float bf2f(unsigned short h) {
  return __uint_as_float(((unsigned)h) << 16);
}

// Build v16bf fragment from two contiguous 8x16-bit halves (elements 0..7, 8..15)
static __device__ __forceinline__ v16bf mk16(v8us lo, v8us hi) {
  U8 ul; ul.u = lo;
  U8 uh; uh.u = hi;
  v16bf f;
#pragma unroll
  for (int i = 0; i < 8; ++i) { f[i] = ul.b[i]; f[i + 8] = uh.b[i]; }
  return f;
}

// Keep 4 smallest, sorted ascending a<=b<=c<=d
static __device__ __forceinline__ void ins4(float v, float& a, float& b, float& c, float& d) {
  if (v < d) {
    d = v;
    if (d < c) { float t = c; c = d; d = t; }
    if (c < b) { float t = b; b = c; c = t; }
    if (b < a) { float t = a; a = b; b = t; }
  }
}

// --------------------------------------------------------------------------
// Prep 1: transpose ref (B,C,M) -> refT (B,M,C) as bf16 hi/lo split.
// LDS-tiled 32x32 transpose: reads coalesced in m, writes coalesced in c.
// --------------------------------------------------------------------------
__global__ __launch_bounds__(256) void ws_prep_transpose(
    const float* __restrict__ ref,
    unsigned short* __restrict__ rtHi,
    unsigned short* __restrict__ rtLo) {
  __shared__ unsigned int tile[32][33];
  const unsigned blk = blockIdx.x;            // 4 * 8 * 128 = 4096 blocks
  const int mTile = blk & 127;                // M/32
  const int cTile = (blk >> 7) & 7;           // C/32
  const int b     = blk >> 10;
  const int m0 = mTile * 32, c0 = cTile * 32;
  const int t  = threadIdx.x;
  const int ml = t & 31;
  const int cg = t >> 5;                      // 0..7
#pragma unroll
  for (int j = 0; j < 4; ++j) {
    const int cl = cg + j * 8;
    const float v = ref[((size_t)(b * CDIM + c0 + cl)) * MDIM + m0 + ml];
    const unsigned short h = f2bf(v);
    const unsigned short l = f2bf(v - bf2f(h));
    tile[cl][ml] = (unsigned)h | ((unsigned)l << 16);
  }
  __syncthreads();
#pragma unroll
  for (int j = 0; j < 4; ++j) {
    const int mlw = (t >> 5) + j * 8;         // m handled by this thread
    const int clw = t & 31;                   // c contiguous across lanes
    const unsigned p = tile[clw][mlw];
    const size_t o = ((size_t)(b * MDIM + m0 + mlw)) * CDIM + c0 + clw;
    rtHi[o] = (unsigned short)p;
    rtLo[o] = (unsigned short)(p >> 16);
  }
}

// --------------------------------------------------------------------------
// Prep 2: per-column ref norms (fp32 exact) + zero the global accumulators.
// --------------------------------------------------------------------------
__global__ __launch_bounds__(256) void ws_prep_rnorm(
    const float* __restrict__ ref,
    float* __restrict__ rnorm,
    float* __restrict__ accum) {
  const int id = blockIdx.x * 256 + threadIdx.x;   // B*M = 16384 threads
  if (id < 2) accum[id] = 0.f;
  const int b = id >> 12;
  const int m = id & (MDIM - 1);
  const float* p = ref + (size_t)b * CDIM * MDIM + m;
  float s = 0.f;
  for (int c = 0; c < CDIM; ++c) { const float v = p[(size_t)c * MDIM]; s += v * v; }
  rnorm[id] = sqrtf(s);
}

// --------------------------------------------------------------------------
// Fused: anchor gather -> bf16 hi/lo GEMM (WMMA) -> cosine dist ->
//        diagonal accumulation + per-row top-4 mining -> global atomics.
// One workgroup (8 wave32) per (batch, 16-row block). Wave w owns column
// tiles w, w+8, ..., keeping running top-4 per (row, column-half) lane.
// --------------------------------------------------------------------------
__global__ __launch_bounds__(256) void fused_gemm_topk(
    const float* __restrict__ sketch,
    const int* __restrict__ ay,
    const int* __restrict__ ax,
    const unsigned short* __restrict__ rtHi,
    const unsigned short* __restrict__ rtLo,
    const float* __restrict__ rnorm,
    float* __restrict__ accum) {
  __shared__ __align__(16) unsigned short AhiS[16][264];   // +8 pad: bank spread
  __shared__ __align__(16) unsigned short AloS[16][264];
  __shared__ float anormS[16];
  __shared__ float distS[8][16][17];
  __shared__ float top4S[8][32][4];
  __shared__ float posS[8][32];

  const int b   = blockIdx.x >> 8;          // 256 row-blocks per batch
  const int n0  = (blockIdx.x & 255) << 4;
  const int tid = threadIdx.x;

  if (tid < 16) anormS[tid] = 0.f;
  __syncthreads();

  // ---- Stage A: gather anchors, bf16 hi/lo split, fp32 sum-of-squares ----
  {
    const int tr = tid & 15;                 // row 0..15
    const int tc = tid >> 4;                 // 16-col group 0..15
    const int n  = n0 + tr;
    const int idx = ay[b * MDIM + n] * WDIM + ax[b * MDIM + n];
    float ss = 0.f;
#pragma unroll
    for (int q = 0; q < 16; ++q) {
      const int c = tc * 16 + q;
      const float v = sketch[((size_t)(b * CDIM + c)) * MDIM + idx];
      ss += v * v;
      const unsigned short h = f2bf(v);
      AhiS[tr][c] = h;
      AloS[tr][c] = f2bf(v - bf2f(h));
    }
    atomicAdd(&anormS[tr], ss);              // ds_add_f32
  }
  __syncthreads();
  if (tid < 16) anormS[tid] = sqrtf(anormS[tid]);
  __syncthreads();

  const int wave = tid >> 5;
  const int lane = tid & 31;
  const int r = lane & 15;                   // A row / C column (N) for lane
  const int g = lane >> 4;                   // lane-half

  // ---- A fragments in registers, reused across all 32 column tiles ----
  // ISA 16-bit A 16x32 layout: lane holds K = 32*kc + 8g + {0..7} (elems 0-7)
  // and K = 32*kc + 16 + 8g + {0..7} (elems 8-15).
  v16bf aHi[8], aLo[8];
#pragma unroll
  for (int kc = 0; kc < 8; ++kc) {
    const int kb = kc * 32;
    aHi[kc] = mk16(*(const v8us*)&AhiS[r][kb + 8 * g],
                   *(const v8us*)&AhiS[r][kb + 16 + 8 * g]);
    aLo[kc] = mk16(*(const v8us*)&AloS[r][kb + 8 * g],
                   *(const v8us*)&AloS[r][kb + 16 + 8 * g]);
  }

  // C/D layout: VGPR j <-> M = j + 8*g, N = r
  float an[8];
#pragma unroll
  for (int j = 0; j < 8; ++j) an[j] = anormS[8 * g + j];

  float t0 = FLT_BIG, t1 = FLT_BIG, t2 = FLT_BIG, t3 = FLT_BIG;
  float posAcc = 0.f;

  const size_t bRowBase = (size_t)b * MDIM * CDIM;

  for (int it = 0; it < 32; ++it) {
    const int mt = wave + (it << 3);         // column tile 0..255
    const int m0 = mt << 4;
    // B 32x16 layout: lane (N = r) holds K = 32*kc + 8*g + {0..7} and
    // K = 32*kc + 16 + 8*g + {0..7} in refT[b][m0+r][c] -> two b128 loads per chunk.
    const size_t rowStart = bRowBase + (size_t)(m0 + r) * CDIM;
    const v8us* bph = (const v8us*)(rtHi + rowStart);
    const v8us* bpl = (const v8us*)(rtLo + rowStart);

    v8f acc = {};
#pragma unroll
    for (int kc = 0; kc < 8; ++kc) {
      const int o = 4 * kc + g;
      const v16bf bHi = mk16(bph[o], bph[o + 2]);
      const v16bf bLo = mk16(bpl[o], bpl[o + 2]);
      // dot = hi*hi' + hi*lo' + lo*hi'  (bf16x3 ~ fp32 accuracy)
      acc = __builtin_amdgcn_wmma_f32_16x16x32_bf16(false, aHi[kc], false, bHi,
                                                    (short)0, acc, false, false);
      acc = __builtin_amdgcn_wmma_f32_16x16x32_bf16(false, aHi[kc], false, bLo,
                                                    (short)0, acc, false, false);
      acc = __builtin_amdgcn_wmma_f32_16x16x32_bf16(false, aLo[kc], false, bHi,
                                                    (short)0, acc, false, false);
    }

    // ---- Epilogue: cosine dist, bounce through per-wave LDS, mine top-4 ----
    const float rn = rnorm[b * MDIM + m0 + r];
#pragma unroll
    for (int j = 0; j < 8; ++j) {
      const float denom = fmaxf(an[j] * rn, EPS);
      distS[wave][j + 8 * g][r] = 1.f - acc[j] / denom;
    }
    // lane (r, g) scans row r, columns 8g..8g+7 (in-wave LDS ordering)
#pragma unroll
    for (int i2 = 0; i2 < 8; ++i2) {
      const int col = 8 * g + i2;
      float v = distS[wave][r][col];
      if (m0 + col == n0 + r) { posAcc += v; v += POS_MASK; }  // diagonal
      ins4(v, t0, t1, t2, t3);
    }
  }

  posS[wave][lane] = posAcc;
  top4S[wave][lane][0] = t0;
  top4S[wave][lane][1] = t1;
  top4S[wave][lane][2] = t2;
  top4S[wave][lane][3] = t3;
  __syncthreads();

  // ---- Merge 16 partial lists per row (8 waves x 2 halves), reduce ----
  if (tid < 16) {
    float a = FLT_BIG, bb = FLT_BIG, c = FLT_BIG, d = FLT_BIG;
    float pos = 0.f;
    for (int w = 0; w < 8; ++w) {
#pragma unroll
      for (int hh = 0; hh < 2; ++hh) {
        const int L = tid + 16 * hh;
        pos += posS[w][L];
#pragma unroll
        for (int k = 0; k < 4; ++k) ins4(top4S[w][L][k], a, bb, c, d);
      }
    }
    const float neg = fmaxf(MARGIN - a, 0.f) + fmaxf(MARGIN - bb, 0.f) +
                      fmaxf(MARGIN - c, 0.f) + fmaxf(MARGIN - d, 0.f);
    atomicAdd(&accum[0], pos);
    atomicAdd(&accum[1], neg);
  }
}

// --------------------------------------------------------------------------
// Final scalar: pos mean over B*M diag entries + neg mean over B*N*4.
// --------------------------------------------------------------------------
__global__ void finalize_k(const float* __restrict__ accum, float* __restrict__ out) {
  out[0] = accum[0] * (1.f / 16384.f) + accum[1] * (1.f / 65536.f);
}

extern "C" void kernel_launch(void* const* d_in, const int* in_sizes, int n_in,
                              void* d_out, int out_size, void* d_ws, size_t ws_size,
                              hipStream_t stream) {
  (void)in_sizes; (void)n_in; (void)out_size; (void)ws_size;
  const float* sketch = (const float*)d_in[0];
  const float* ref    = (const float*)d_in[1];
  const int*   ay     = (const int*)d_in[2];
  const int*   ax     = (const int*)d_in[3];

  char* ws = (char*)d_ws;                              // needs ~16.9 MB
  unsigned short* rtHi = (unsigned short*)(ws);                          // 8 MB
  unsigned short* rtLo = (unsigned short*)(ws + (size_t)8 * 1024 * 1024); // 8 MB
  float* rnorm = (float*)(ws + (size_t)16 * 1024 * 1024);                // 64 KB
  float* accum = (float*)(ws + (size_t)16 * 1024 * 1024 + 64 * 1024);    // 8 B

  ws_prep_transpose<<<4096, 256, 0, stream>>>(ref, rtHi, rtLo);
  ws_prep_rnorm<<<64, 256, 0, stream>>>(ref, rnorm, accum);
  fused_gemm_topk<<<BATCH * (MDIM / 16), 256, 0, stream>>>(
      sketch, ay, ax, rtHi, rtLo, rnorm, accum);
  finalize_k<<<1, 1, 0, stream>>>(accum, (float*)d_out);
}
